// EBMb_17360257810670
// MI455X (gfx1250) — matrix operand order repaired and checked by reading, not compiled, not run on hardware
//
#include <hip/hip_runtime.h>
#include <hip/hip_bf16.h>
#include <cstdint>
#include <cstddef>

// ---------------------------------------------------------------------------
// Persistent fused Langevin-MCMC sampler for the 2-128-128-128-128-1 SiLU EBM.
// One kernel runs all 60 steps. Weights are pre-fragmented into LDS in the
// exact CDNA5 WMMA 16-bit A/B lane layouts; all GEMMs use
// v_wmma_f32_16x16x32_f16 with f32 accumulation. Sigmoids use the native
// v_exp_f32 / v_rcp_f32 transcendentals (co-execute with XDL WMMA); all
// staging epilogues run in packed f16 (v_cvt_pk_f16_f32 / v_pk_mul_f16).
// ---------------------------------------------------------------------------

typedef _Float16 h16;
typedef __attribute__((ext_vector_type(16))) _Float16 v16h;
typedef __attribute__((ext_vector_type(8)))  _Float16 v8h;
typedef __attribute__((ext_vector_type(2)))  _Float16 v2h;
typedef __attribute__((ext_vector_type(8)))  float    v8f;

#define NSTEPS 60
#define HID    128
#define STR    136          // padded LDS row stride in halves (bank spread)
#define WPB    8            // waves per block (wave32)
#define CPB    128          // chains per block = WPB * 16

// ---- LDS byte layout (dynamic shared, ~241.5 KB, WGP has 320 KB) ----
#define OFF_FWD   0          // 3 mats * 32 frags * 32 lanes * 16 halves * 2B = 98304
#define OFF_BWD   98304      // transposed fragments, same size
#define OFF_W1F   196608     // 8 frags  * 1024B = 8192   (w1 fwd, K zero-padded)
#define OFF_W1B   204800     // 4 frags  * 1024B = 4096   (w1^T,  N zero-padded)
#define OFF_BIAS  208896     // 4 * 128 f32 = 2048
#define OFF_W5    210944     // 128 f32 = 512
#define OFF_STG   211456     // 8 waves * 16*STR*2 = 34816
#define OFF_DX    246272     // 8 waves * 32 f32 = 1024
#define SMEM_BYTES 247296

// Same-wave LDS ops are processed in order by the DS pipeline (ISA: DScnt ops
// are in-order per wave), so phase boundaries only need a *compiler* reorder
// barrier; the backend inserts precise partial s_wait_dscnt for register deps.
__device__ __forceinline__ void sched_fence() {
  asm volatile("" ::: "memory");
}

__device__ __forceinline__ v8f wmma16(v16h a, v16h b, v8f c) {
  return __builtin_amdgcn_wmma_f32_16x16x32_f16(false, a, false, b,
                                                (short)0, c, false, false);
}

// A operand (16x32 f16): lane holds row M=lane%16; halves 0-7 = K base..base+7,
// halves 8-15 = K base+16..base+23, base = kt*32 + 8*(lane/16).
__device__ __forceinline__ v16h loadA(const h16* stg, int m, int kbase) {
  v8h lo = *(const v8h*)(stg + m * STR + kbase);
  v8h hi = *(const v8h*)(stg + m * STR + kbase + 16);
  return __builtin_shufflevector(lo, hi, 0,1,2,3,4,5,6,7,8,9,10,11,12,13,14,15);
}

// B operand (32x16 f16): pre-fragmented, lane's 16 halves stored contiguously.
__device__ __forceinline__ v16h loadB(const h16* fragbase, int lane) {
  return *(const v16h*)(fragbase + lane * 16);
}

// Fast SiLU: native v_exp_f32 (2^x) + v_rcp_f32; silu' reuses h = z*sig:
// silu'(z) = sig + z*sig*(1-sig) = fma(h, 1-sig, sig).
__device__ __forceinline__ void fast_silu(float z, float& h, float& sp) {
  float e   = __builtin_amdgcn_exp2f(z * -1.44269504088896341f);  // exp(-z)
  float sig = __builtin_amdgcn_rcpf(1.f + e);
  h  = z * sig;
  sp = __builtin_fmaf(h, 1.f - sig, sig);
}

__device__ __forceinline__ v2h cvt2h(float a, float b) {
  v2h r; r.x = (h16)a; r.y = (h16)b; return r;      // v_cvt_pk_f16_f32
}
__device__ __forceinline__ uint32_t as_u32(v2h v) {
  union { v2h v; uint32_t u; } x; x.v = v; return x.u;
}
__device__ __forceinline__ v2h as_v2h(uint32_t u) {
  union { uint32_t u; v2h v; } x; x.u = u; return x.v;
}

// Store packed pair into rows m and m+1 of the staging tile (same column):
// ds_store_b16 (low half) + ds_store_b16_d16_hi (high half).
__device__ __forceinline__ void store_pair(h16* base, int addr, uint32_t packed) {
  union { uint32_t u; h16 h[2]; } q; q.u = packed;
  base[addr]       = q.h[0];
  base[addr + STR] = q.h[1];
}

__global__ __launch_bounds__(256, 1)
void ebm_langevin_kernel(const float* __restrict__ x0g,
                         const float* __restrict__ w1, const float* __restrict__ b1,
                         const float* __restrict__ w2, const float* __restrict__ b2,
                         const float* __restrict__ w3, const float* __restrict__ b3,
                         const float* __restrict__ w4, const float* __restrict__ b4,
                         const float* __restrict__ w5, const float* __restrict__ noise,
                         float* __restrict__ out, int NTOT)
{
  extern __shared__ float4 smem_v[];
  char* smem = (char*)smem_v;
  h16*   fwdF  = (h16*)(smem + OFF_FWD);   // B frags of W2,W3,W4
  h16*   bwdF  = (h16*)(smem + OFF_BWD);   // B frags of W2^T,W3^T,W4^T
  h16*   w1F   = (h16*)(smem + OFF_W1F);
  h16*   w1B   = (h16*)(smem + OFF_W1B);
  float* biasL = (float*)(smem + OFF_BIAS);
  float* w5L   = (float*)(smem + OFF_W5);

  const int tid  = threadIdx.x;
  const int wave = tid >> 5;
  const int lane = tid & 31;
  const int lm   = lane & 15;   // N index inside a tile (C/D layout), row M for A
  const int lh   = lane >> 4;   // selects M+8 half (C/D) / K half (A,B)

  h16*   stg = (h16*)(smem + OFF_STG) + wave * (16 * STR);  // per-wave 16x128 f16
  float* dxb = (float*)(smem + OFF_DX) + wave * 32;         // per-wave 16x2 f32

  // ------------------ init: pre-fragment all weights into LDS ------------------
  {
    const float* Wsrc[3] = { w2, w3, w4 };
    for (int g = tid; g < 6 * 1024; g += 256) {
      int md  = g >> 10;           // 0..2 fwd W2..W4, 3..5 bwd (transposed)
      int rem = g & 1023;
      int f   = rem >> 5;          // frag = kt*8 + nt
      int L   = rem & 31;
      int kt = f >> 3, nt = f & 7;
      int hi = L >> 4, nl = L & 15;
      bool bwd = md >= 3;
      const float* W = Wsrc[bwd ? md - 3 : md];
      h16* dst = (bwd ? bwdF + (md - 3) * 16384 : fwdF + md * 16384)
                 + f * 512 + L * 16;
      int n  = nt * 16 + nl;       // B column (lane)
      int kb = kt * 32 + hi * 8;   // B row base (half i -> K kb + i + (i&8))
      if (!bwd) {
        #pragma unroll
        for (int i = 0; i < 16; ++i) dst[i] = (h16)W[(kb + i + (i & 8)) * HID + n];
      } else {                     // B = W^T  => B[k][n] = W[n][k]
        #pragma unroll
        for (int i = 0; i < 16; ++i) dst[i] = (h16)W[n * HID + kb + i + (i & 8)];
      }
    }
    // w1 forward fragment: B[k][n] nonzero only for k=0,1 (K padded to 32)
    for (int g = tid; g < 8 * 32; g += 256) {
      int nt = g >> 5, L = g & 31;
      h16* dst = w1F + nt * 512 + L * 16;
      #pragma unroll
      for (int i = 0; i < 16; ++i) dst[i] = (h16)0.f;
      if (L < 16) {                // lanes 0..15 hold K=0..7,16..23
        dst[0] = (h16)w1[0 * HID + nt * 16 + L];
        dst[1] = (h16)w1[1 * HID + nt * 16 + L];
      }
    }
    // w1^T fragment: B[k][n] nonzero only for n=0,1 (N padded to 16)
    for (int g = tid; g < 4 * 32; g += 256) {
      int kt = g >> 5, L = g & 31;
      int hi = L >> 4, nl = L & 15;
      h16* dst = w1B + kt * 512 + L * 16;
      #pragma unroll
      for (int i = 0; i < 16; ++i)
        dst[i] = (nl < 2) ? (h16)w1[nl * HID + kt * 32 + hi * 8 + i + (i & 8)] : (h16)0.f;
    }
    for (int g = tid; g < 512; g += 256) {
      int l = g >> 7, n = g & 127;
      const float* bsrc = (l == 0) ? b1 : (l == 1) ? b2 : (l == 2) ? b3 : b4;
      biasL[g] = bsrc[n];
    }
    for (int g = tid; g < HID; g += 256) w5L[g] = w5[g];
  }
  __syncthreads();

  // ------------------ per-chain state: lanes 0..15 own one chain each ----------
  const int  chainBase = blockIdx.x * CPB + wave * 16;
  const int  myChain   = chainBase + lm;
  const bool owner     = (lane < 16) && (myChain < NTOT);
  float xr = 0.f, yr = 0.f;
  if (owner) {
    xr = x0g[2 * (size_t)myChain + 0];
    yr = x0g[2 * (size_t)myChain + 1];
  }

  uint32_t sd[3][32];   // silu'(z_l) for l=1..3, fp16x2 packed, D-tile layout.
                        // silu'(z4) is consumed in-place by the fused g4 seed.

  #pragma unroll 1
  for (int step = 0; step < NSTEPS; ++step) {
    const float eps = 10.f * (1.f - (float)step / (float)NSTEPS);
    const float nco = sqrtf(2.f * eps) * 0.005f;

    // ===== layer 1 : z1 = [x,y] @ w1 + b1  (K zero-padded to 32) =====
    v16h a1 = {};
    a1[0] = (h16)xr;               // only K=0,1 matter; padded B rows are zero
    a1[1] = (h16)yr;
    #pragma unroll
    for (int nt = 0; nt < 8; ++nt) {
      v8f acc = {};
      acc = wmma16(a1, loadB(w1F + nt * 512, lane), acc);
      const float bn = biasL[0 * HID + nt * 16 + lm];
      #pragma unroll
      for (int rp = 0; rp < 4; ++rp) {
        float h0, h1, p0, p1;
        fast_silu(acc[2 * rp]     + bn, h0, p0);
        fast_silu(acc[2 * rp + 1] + bn, h1, p1);
        store_pair(stg, (2 * rp + 8 * lh) * STR + nt * 16 + lm,
                   as_u32(cvt2h(h0, h1)));
        sd[0][nt * 4 + rp] = as_u32(cvt2h(p0, p1));
      }
    }

    // ===== layers 2..4 : z = h @ W + b =====
    // Layers 2,3: stage silu(z), keep silu'(z) in regs.
    // Layer 4 (li==2): fuse the backward seed -> stage g4 = w5 ⊙ silu'(z4)
    // directly (h4 is never consumed; dE/dh4 is the constant w5 vector).
    #pragma unroll
    for (int li = 0; li < 3; ++li) {
      sched_fence();
      v16h Af[4];
      #pragma unroll
      for (int kt = 0; kt < 4; ++kt) Af[kt] = loadA(stg, lm, kt * 32 + 8 * lh);
      sched_fence();
      const h16* fb = fwdF + li * 16384;
      #pragma unroll
      for (int nt = 0; nt < 8; ++nt) {
        v8f acc = {};
        #pragma unroll
        for (int kt = 0; kt < 4; ++kt)
          acc = wmma16(Af[kt], loadB(fb + (kt * 8 + nt) * 512, lane), acc);
        const float bn = biasL[(li + 1) * HID + nt * 16 + lm];
        v2h wnh = {};
        if (li == 2) {
          const float wn = w5L[nt * 16 + lm];
          wnh = cvt2h(wn, wn);
        }
        #pragma unroll
        for (int rp = 0; rp < 4; ++rp) {
          float h0, h1, p0, p1;
          fast_silu(acc[2 * rp]     + bn, h0, p0);
          fast_silu(acc[2 * rp + 1] + bn, h1, p1);
          const int addr = (2 * rp + 8 * lh) * STR + nt * 16 + lm;
          if (li != 2) {
            store_pair(stg, addr, as_u32(cvt2h(h0, h1)));
            sd[li + 1][nt * 4 + rp] = as_u32(cvt2h(p0, p1));
          } else {     // g4 seed: w5 ⊙ silu'(z4), packed f16 multiply
            store_pair(stg, addr, as_u32(cvt2h(p0, p1) * wnh));
          }
        }
      }
    }

    // ===== backward: g_l = (g_{l+1} @ W_{l+1}^T) ⊙ silu'(z_l) =====
    #pragma unroll
    for (int bi = 0; bi < 3; ++bi) {
      const int wsel = 2 - bi;     // W4^T, W3^T, W2^T ; pairs with sd[2],sd[1],sd[0]
      sched_fence();
      v16h Ab[4];
      #pragma unroll
      for (int kt = 0; kt < 4; ++kt) Ab[kt] = loadA(stg, lm, kt * 32 + 8 * lh);
      sched_fence();
      const h16* bb = bwdF + wsel * 16384;
      #pragma unroll
      for (int nt = 0; nt < 8; ++nt) {
        v8f acc = {};
        #pragma unroll
        for (int kt = 0; kt < 4; ++kt)
          acc = wmma16(Ab[kt], loadB(bb + (kt * 8 + nt) * 512, lane), acc);
        #pragma unroll
        for (int rp = 0; rp < 4; ++rp) {
          // packed: one v_cvt_pk_f16_f32 + one v_pk_mul_f16 per pair
          v2h t = cvt2h(acc[2 * rp], acc[2 * rp + 1]) * as_v2h(sd[wsel][nt * 4 + rp]);
          store_pair(stg, (2 * rp + 8 * lh) * STR + nt * 16 + lm, as_u32(t));
        }
      }
    }

    // ===== dE/dx = g1 @ W1^T  (N zero-padded to 16; cols 0,1 valid) =====
    sched_fence();
    v8f dacc = {};
    #pragma unroll
    for (int kt = 0; kt < 4; ++kt)
      dacc = wmma16(loadA(stg, lm, kt * 32 + 8 * lh), loadB(w1B + kt * 512, lane), dacc);
    sched_fence();
    if (lm < 2) {                  // lanes 0,1,16,17 hold the valid D columns
      #pragma unroll
      for (int r = 0; r < 8; ++r) {
        float g = dacc[r];
        g = fminf(fmaxf(g, -0.03f), 0.03f);
        dxb[(r + 8 * lh) * 2 + lm] = g;
      }
    }
    sched_fence();

    // ===== Langevin update (chain owners only) =====
    if (owner) {
      const float gx = dxb[lm * 2 + 0];
      const float gy = dxb[lm * 2 + 1];
      const float2 nz = *((const float2*)noise + (size_t)step * (size_t)NTOT + myChain);
      xr = fminf(fmaxf(xr + nco * nz.x + eps * gx, -2.43f), 3.05f);
      yr = fminf(fmaxf(yr + nco * nz.y + eps * gy, -2.43f), 3.05f);
    }
    sched_fence();
  }

  if (owner) {
    ((float2*)out)[myChain] = make_float2(xr, yr);
  }
}

extern "C" void kernel_launch(void* const* d_in, const int* in_sizes, int n_in,
                              void* d_out, int out_size, void* d_ws, size_t ws_size,
                              hipStream_t stream) {
  (void)n_in; (void)out_size; (void)d_ws; (void)ws_size;
  const float* x0    = (const float*)d_in[0];
  const float* w1    = (const float*)d_in[1];
  const float* b1    = (const float*)d_in[2];
  const float* w2    = (const float*)d_in[3];
  const float* b2    = (const float*)d_in[4];
  const float* w3    = (const float*)d_in[5];
  const float* b3    = (const float*)d_in[6];
  const float* w4    = (const float*)d_in[7];
  const float* b4    = (const float*)d_in[8];
  const float* w5    = (const float*)d_in[9];
  // d_in[10] = b5: constant offset, does not affect dE/dx
  const float* noise = (const float*)d_in[11];
  float* out = (float*)d_out;

  const int N = in_sizes[0] / 2;
  (void)hipFuncSetAttribute(reinterpret_cast<const void*>(ebm_langevin_kernel),
                            hipFuncAttributeMaxDynamicSharedMemorySize, SMEM_BYTES);
  const int grid = (N + CPB - 1) / CPB;
  ebm_langevin_kernel<<<grid, 256, SMEM_BYTES, stream>>>(
      x0, w1, b1, w2, b2, w3, b3, w4, b4, w5, noise, out, N);
}
